// AudioDecoder_35296041239275
// MI455X (gfx1250) — matrix operand order repaired and checked by reading, not compiled, not run on hardware
//
#include <hip/hip_runtime.h>
#include <hip/hip_bf16.h>

// ---------------------------------------------------------------------------
// MI455X (gfx1250) bf16 WMMA pipeline for the 2-layer attention+conv block.
// GEMM: C(MxN) = A(MxK) * B, with B supplied in (N,K) row-major ("NT") layout
// so every LDS fragment read is a contiguous ds_load_b128.
// ---------------------------------------------------------------------------

typedef __attribute__((ext_vector_type(16))) __bf16 v16bf;
typedef __attribute__((ext_vector_type(8)))  float  v8f;
typedef __attribute__((ext_vector_type(4)))  int    vi4;   // vector int4 (builtin param type)

#define BB   4
#define TT   1024
#define HIDN 1024
#define NHN  16
#define NKVN 8
#define GN   2
#define HDN  64
#define RANKN 256
#define FFN  4096
#define KTAP 9
#define TP   (TT + 8)   // padded sequence length for K=9 conv (pad 4 each side)

#if defined(__gfx1250__) && __has_builtin(__builtin_amdgcn_global_load_async_to_lds_b128) && \
    __has_builtin(__builtin_amdgcn_s_wait_asynccnt)
#define USE_ASYNC 1
#else
#define USE_ASYNC 0
#endif

__device__ __forceinline__ unsigned short f2bf(float f) {
  union { float f; unsigned int u; } x; x.f = f;
  unsigned int u = x.u;
  u += 0x7FFFu + ((u >> 16) & 1u);   // round-to-nearest-even
  return (unsigned short)(u >> 16);
}

#if USE_ASYNC
__device__ __forceinline__ void async_b128(const unsigned short* g, unsigned short* l) {
  __builtin_amdgcn_global_load_async_to_lds_b128(
      (vi4*)g,                                        // global src (generic vector ptr)
      (__attribute__((address_space(3))) vi4*)l,      // LDS dst
      0 /*imm offset*/, 0 /*cpol*/);
}
#endif

// ---------------------------------------------------------------------------
// Tiled bf16 GEMM:  C(MxN, f32|bf16) = alpha*A(MxK)*B + beta*C
//   A  : (M,K) row-major bf16
//   BT : (N,K) row-major bf16  (i.e. B transposed storage)
// Caller guarantees: M%128==0, N%64==0, K%32==0, bases 16B aligned,
// lda/ldbt multiples of 8 elements.
// Block = 256 threads = 8 waves (4x2). Each wave owns a 32x32 output tile ->
// 4 x v_wmma_f32_16x16x32_bf16 per 32-K step. Double-buffered LDS staging via
// gfx1250 GLOBAL_LOAD_ASYNC_TO_LDS_B128 (sync fallback if builtin missing).
// A rows are stored in LDS with 8-elem K-groups permuted {0,2,1,3} so each
// half-wave's 16 fragment elements are contiguous (2 x ds_load_b128).
// ---------------------------------------------------------------------------
__global__ __launch_bounds__(256)
void k_gemm_bf16(const unsigned short* __restrict__ A, int lda,
                 const unsigned short* __restrict__ BT, int ldbt,
                 void* __restrict__ Cp, int ldc,
                 int M, int N, int K,
                 float alpha, float beta, int c_is_bf16) {
  __shared__ unsigned short As[2][128][32];   // 16 KB, K-group-permuted rows
  __shared__ unsigned short Bs[2][64][32];    //  8 KB, straight (n,k) rows

  const int tid  = threadIdx.x;
  const int wave = tid >> 5;
  const int lane = tid & 31;
  const int wm   = wave & 3;            // 4 row-waves  (32 rows each)
  const int wn   = wave >> 2;           // 2 col-waves  (32 cols each)
  const int l15  = lane & 15;
  const int hi   = lane >> 4;

  const int bm = blockIdx.x * 128;
  const int bn = blockIdx.y * 64;

  // staging coordinates: A tile 128x32 (2 x b128/thread), B tile 64x32 (1)
  const int ar  = tid >> 1;             // 0..127
  const int ag  = (tid & 1) * 2;        // K-groups {0,1} or {2,3}
  const int pA0 = (tid & 1) ? 1 : 0;    // perm {0,2,1,3}
  const int pA1 = (tid & 1) ? 3 : 2;
  const int br  = tid >> 2;             // 0..63
  const int bg  = tid & 3;

  const unsigned short* Aptr = A  + (size_t)(bm + ar) * lda  + ag * 8;
  const unsigned short* Bptr = BT + (size_t)(bn + br) * ldbt + bg * 8;

  v8f acc00 = {}, acc01 = {}, acc10 = {}, acc11 = {};

  auto stage = [&](int buf, int k0) {
#if USE_ASYNC
    async_b128(Aptr + k0,     &As[buf][ar][pA0 * 8]);
    async_b128(Aptr + k0 + 8, &As[buf][ar][pA1 * 8]);
    async_b128(Bptr + k0,     &Bs[buf][br][bg * 8]);
#else
    const uint4 va0 = *(const uint4*)(Aptr + k0);
    const uint4 va1 = *(const uint4*)(Aptr + k0 + 8);
    const uint4 vb  = *(const uint4*)(Bptr + k0);
    *(uint4*)&As[buf][ar][pA0 * 8] = va0;
    *(uint4*)&As[buf][ar][pA1 * 8] = va1;
    *(uint4*)&Bs[buf][br][bg * 8]  = vb;
#endif
  };

  const int nk = K >> 5;
  stage(0, 0);
  for (int kt = 0; kt < nk; ++kt) {
    const int buf = kt & 1;
    if (kt + 1 < nk) {
      stage(buf ^ 1, (kt + 1) * 32);
#if USE_ASYNC
      __builtin_amdgcn_s_wait_asynccnt(3);   // previous tile's 3 ops complete
#endif
    } else {
#if USE_ASYNC
      __builtin_amdgcn_s_wait_asynccnt(0);
#endif
    }
    __syncthreads();   // tile `buf` visible to all waves

    union Fr { v16bf v; uint4 q[2]; } a0, a1, b0, b1;
    const int am = wm * 32 + l15;
    a0.q[0] = *(const uint4*)&As[buf][am][hi * 16];
    a0.q[1] = *(const uint4*)&As[buf][am][hi * 16 + 8];
    a1.q[0] = *(const uint4*)&As[buf][am + 16][hi * 16];
    a1.q[1] = *(const uint4*)&As[buf][am + 16][hi * 16 + 8];
    const int bn0 = wn * 32 + l15;
    b0.q[0] = *(const uint4*)&Bs[buf][bn0][hi * 8];
    b0.q[1] = *(const uint4*)&Bs[buf][bn0][16 + hi * 8];
    b1.q[0] = *(const uint4*)&Bs[buf][bn0 + 16][hi * 8];
    b1.q[1] = *(const uint4*)&Bs[buf][bn0 + 16][16 + hi * 8];

    acc00 = __builtin_amdgcn_wmma_f32_16x16x32_bf16(false, a0.v, false, b0.v,
                                                    (short)0, acc00, false, false);
    acc01 = __builtin_amdgcn_wmma_f32_16x16x32_bf16(false, a0.v, false, b1.v,
                                                    (short)0, acc01, false, false);
    acc10 = __builtin_amdgcn_wmma_f32_16x16x32_bf16(false, a1.v, false, b0.v,
                                                    (short)0, acc10, false, false);
    acc11 = __builtin_amdgcn_wmma_f32_16x16x32_bf16(false, a1.v, false, b1.v,
                                                    (short)0, acc11, false, false);
    __syncthreads();   // safe to overwrite tile `buf` next round
  }

  // C/D layout: VGPR r holds row m=r+8*hi, column n=l15 of each 16x16 tile
  const int rowb = bm + wm * 32 + 8 * hi;
  const int colb = bn + wn * 32 + l15;
#pragma unroll
  for (int mt = 0; mt < 2; ++mt) {
    const v8f* am0 = (mt == 0) ? &acc00 : &acc10;
    const v8f* am1 = (mt == 0) ? &acc01 : &acc11;
#pragma unroll
    for (int r = 0; r < 8; ++r) {
      const int row = rowb + mt * 16 + r;
      const float v0 = alpha * (*am0)[r];
      const float v1 = alpha * (*am1)[r];
      if (c_is_bf16) {
        unsigned short* C = (unsigned short*)Cp;
        C[(size_t)row * ldc + colb]      = f2bf(v0);
        C[(size_t)row * ldc + colb + 16] = f2bf(v1);
      } else {
        float* C = (float*)Cp;
        const size_t i0 = (size_t)row * ldc + colb;
        const size_t i1 = i0 + 16;
        const float o0 = (beta != 0.f) ? beta * C[i0] : 0.f;
        const float o1 = (beta != 0.f) ? beta * C[i1] : 0.f;
        C[i0] = o0 + v0;
        C[i1] = o1 + v1;
      }
    }
  }
}

// ---------------------------------------------------------------------------
// LayerNorm over C features of one row per block; outputs bf16.
// ---------------------------------------------------------------------------
__global__ __launch_bounds__(256)
void k_layernorm_cast(const float* __restrict__ x, int row_stride, int C,
                      const float* __restrict__ w, const float* __restrict__ b,
                      unsigned short* __restrict__ out, int out_stride) {
  const int row = blockIdx.x;
  const float* xr = x + (size_t)row * row_stride;
  float s = 0.f, ss = 0.f;
  for (int c = threadIdx.x; c < C; c += blockDim.x) {
    const float v = xr[c];
    s += v; ss += v * v;
  }
  __shared__ float rs[8], rss[8];
  for (int o = 16; o > 0; o >>= 1) {
    s  += __shfl_down(s, o, 32);
    ss += __shfl_down(ss, o, 32);
  }
  const int lane = threadIdx.x & 31, wv = threadIdx.x >> 5;
  if (lane == 0) { rs[wv] = s; rss[wv] = ss; }
  __syncthreads();
  if (threadIdx.x == 0) {
    float ts = 0.f, tss = 0.f;
    for (int i = 0; i < 8; ++i) { ts += rs[i]; tss += rss[i]; }
    const float mean = ts / C;
    rs[0]  = mean;
    rss[0] = tss / C - mean * mean;
  }
  __syncthreads();
  const float mean = rs[0];
  const float inv  = rsqrtf(rss[0] + 1e-5f);
  for (int c = threadIdx.x; c < C; c += blockDim.x) {
    const float v = (xr[c] - mean) * inv * w[c] + b[c];
    out[(size_t)row * out_stride + c] = f2bf(v);
  }
}

// ---------------------------------------------------------------------------
// Row softmax of S (scaled), cast to bf16 P. One row per block, N columns.
// ---------------------------------------------------------------------------
__global__ __launch_bounds__(256)
void k_softmax_bf16(const float* __restrict__ S, unsigned short* __restrict__ P,
                    int N, float scale) {
  const int row = blockIdx.x;
  const float* sr = S + (size_t)row * N;
  __shared__ float red[8], red2[8];
  const int lane = threadIdx.x & 31, wv = threadIdx.x >> 5;

  float mx = -3.0e38f;
  for (int c = threadIdx.x; c < N; c += blockDim.x) mx = fmaxf(mx, sr[c] * scale);
  for (int o = 16; o > 0; o >>= 1) mx = fmaxf(mx, __shfl_xor(mx, o, 32));
  if (lane == 0) red[wv] = mx;
  __syncthreads();
  if (threadIdx.x == 0) {
    float m = red[0];
    for (int i = 1; i < 8; ++i) m = fmaxf(m, red[i]);
    red[0] = m;
  }
  __syncthreads();
  mx = red[0];

  float sum = 0.f;
  for (int c = threadIdx.x; c < N; c += blockDim.x) sum += __expf(sr[c] * scale - mx);
  for (int o = 16; o > 0; o >>= 1) sum += __shfl_down(sum, o, 32);
  if (lane == 0) red2[wv] = sum;
  __syncthreads();
  if (threadIdx.x == 0) {
    float t = 0.f;
    for (int i = 0; i < 8; ++i) t += red2[i];
    red2[0] = t;
  }
  __syncthreads();
  const float inv = 1.f / red2[0];
  for (int c = threadIdx.x; c < N; c += blockDim.x)
    P[(size_t)row * N + c] = f2bf(__expf(sr[c] * scale - mx) * inv);
}

// ---------------------------------------------------------------------------
// RoPE + repack.  inv_freq = 10000^(-2d/HD).
// q  -> qbf  (B,NH,T,HD)          (GEMM A operand for scores)
// key-> kbf  (B,NKV,T,HD)         (NT layout for scores GEMM: (N=t',K=d))
// val-> vbfT (B,NKV,HD,T)         (NT layout for PV GEMM:     (N=d,K=t'))
// ---------------------------------------------------------------------------
__global__ __launch_bounds__(256)
void k_rope_pack_q(const float* __restrict__ q, unsigned short* __restrict__ qbf) {
  const int idx = blockIdx.x * blockDim.x + threadIdx.x;
  if (idx >= BB * TT * NHN * (HDN / 2)) return;
  const int d = idx & 31;
  const int h = (idx >> 5) & (NHN - 1);
  const int t = (idx >> 9) & (TT - 1);
  const int b = idx >> 19;
  const float inv = __expf(-(float)(2 * d) * (9.210340371976184f / (float)HDN));
  const float ang = (float)t * inv;
  const float c = __cosf(ang), s = __sinf(ang);
  const size_t src = ((size_t)(b * TT + t)) * HIDN + h * HDN + d;
  const float x1 = q[src], x2 = q[src + 32];
  const size_t dst = (((size_t)(b * NHN + h)) * TT + t) * HDN + d;
  qbf[dst]      = f2bf(x1 * c - x2 * s);
  qbf[dst + 32] = f2bf(x2 * c + x1 * s);
}

__global__ __launch_bounds__(256)
void k_rope_pack_kv(const float* __restrict__ kv,
                    unsigned short* __restrict__ kbf,    // (B,NKV,T,HD)
                    unsigned short* __restrict__ vbfT) { // (B,NKV,HD,T)
  const int idx = blockIdx.x * blockDim.x + threadIdx.x;
  if (idx >= BB * TT * NKVN * (HDN / 2)) return;
  const int d = idx & 31;
  const int h = (idx >> 5) & (NKVN - 1);
  const int t = (idx >> 8) & (TT - 1);
  const int b = idx >> 18;
  const float inv = __expf(-(float)(2 * d) * (9.210340371976184f / (float)HDN));
  const float ang = (float)t * inv;
  const float c = __cosf(ang), s = __sinf(ang);
  const size_t row = (size_t)(b * TT + t);
  // key (cols 0..511): rope, natural layout
  const size_t ks = row * HIDN + h * HDN + d;
  const float x1 = kv[ks], x2 = kv[ks + 32];
  const size_t kd = (((size_t)(b * NKVN + h)) * TT + t) * HDN + d;
  kbf[kd]      = f2bf(x1 * c - x2 * s);
  kbf[kd + 32] = f2bf(x2 * c + x1 * s);
  // value (cols 512..1023): plain cast, transposed (d-major) for PV GEMM
  const size_t vs = row * HIDN + NKVN * HDN + h * HDN + d;
  const size_t vd = (((size_t)(b * NKVN + h)) * HDN + d) * TT + t;
  vbfT[vd]           = f2bf(kv[vs]);
  vbfT[vd + 32 * TT] = f2bf(kv[vs + 32]);
}

// ---------------------------------------------------------------------------
// Weight prep. Projection weights W (O,C) row-major are ALREADY the (N,K)
// layout the GEMM wants -> pure cast. Conv weights (O,C,Ktap) -> per-tap
// (N=O, K=C): out[(k*O + o)*C + c] = W[(o*C + c)*Ktap + k].
// ---------------------------------------------------------------------------
__global__ __launch_bounds__(256)
void k_cast_bf16(const float* __restrict__ W, unsigned short* __restrict__ out, int n) {
  const int i = blockIdx.x * blockDim.x + threadIdx.x;
  if (i < n) out[i] = f2bf(W[i]);
}

__global__ __launch_bounds__(256)
void k_conv_repack(const float* __restrict__ W, unsigned short* __restrict__ out,
                   int O, int C, int Kt) {
  const long idx = (long)blockIdx.x * blockDim.x + threadIdx.x;
  if (idx >= (long)O * C * Kt) return;
  const int c = (int)(idx % C);
  const int o = (int)((idx / C) % O);
  const int k = (int)(idx / ((long)C * O));
  out[idx] = f2bf(W[((size_t)o * C + c) * Kt + k]);
}

// ---------------------------------------------------------------------------
// Padding / activation / residual helpers
// ---------------------------------------------------------------------------
__global__ __launch_bounds__(256)
void k_pad_bf16(const unsigned short* __restrict__ src, unsigned short* __restrict__ dst,
                int C) {   // src (BB*TT, C) -> dst (BB, TP, C) zero-padded
  const long idx = (long)blockIdx.x * blockDim.x + threadIdx.x;
  if (idx >= (long)BB * TP * C) return;
  const int  c = (int)(idx % C);
  const long r = idx / C;
  const int  t = (int)(r % TP);
  const int  b = (int)(r / TP);
  unsigned short v = 0;
  if (t >= 4 && t < TT + 4) v = src[((size_t)(b * TT + (t - 4))) * C + c];
  dst[idx] = v;
}

__global__ __launch_bounds__(256)
void k_relu_bias_pad(const float* __restrict__ src, const float* __restrict__ bias,
                     unsigned short* __restrict__ dst) { // (BB*TT,FFN)->(BB,TP,FFN)
  const long idx = (long)blockIdx.x * blockDim.x + threadIdx.x;
  if (idx >= (long)BB * TP * FFN) return;
  const int  c = (int)(idx % FFN);
  const long r = idx / FFN;
  const int  t = (int)(r % TP);
  const int  b = (int)(r / TP);
  unsigned short v = 0;
  if (t >= 4 && t < TT + 4) {
    const float f = src[((size_t)(b * TT + (t - 4))) * FFN + c] + bias[c];
    v = f2bf(fmaxf(f, 0.f));
  }
  dst[idx] = v;
}

__global__ __launch_bounds__(256)
void k_add_res(float* __restrict__ x, const float* __restrict__ y, int n) {
  const int i = blockIdx.x * blockDim.x + threadIdx.x;
  if (i < n) x[i] += y[i];
}

__global__ __launch_bounds__(256)
void k_bias_add_res(float* __restrict__ x, const float* __restrict__ y,
                    const float* __restrict__ bias, int n) {
  const int i = blockIdx.x * blockDim.x + threadIdx.x;
  if (i < n) x[i] += y[i] + bias[i & (HIDN - 1)];
}

// ---------------------------------------------------------------------------
// Host-side orchestration
// ---------------------------------------------------------------------------
static inline void gemm(hipStream_t st, const void* A, int lda, const void* BT, int ldbt,
                        void* C, int ldc, int M, int N, int K,
                        float alpha, float beta, int c_bf16) {
  dim3 grid(M / 128, N / 64);
  k_gemm_bf16<<<grid, dim3(256), 0, st>>>((const unsigned short*)A, lda,
                                          (const unsigned short*)BT, ldbt,
                                          C, ldc, M, N, K, alpha, beta, c_bf16);
}

extern "C" void kernel_launch(void* const* d_in, const int* in_sizes, int n_in,
                              void* d_out, int out_size, void* d_ws, size_t ws_size,
                              hipStream_t stream) {
  (void)in_sizes; (void)n_in; (void)out_size; (void)ws_size;
  const float* in_x = (const float*)d_in[0];
  const float* anw  = (const float*)d_in[1];
  const float* anb  = (const float*)d_in[2];
  const float* qw   = (const float*)d_in[3];
  const float* kvaw = (const float*)d_in[4];
  const float* kvnw = (const float*)d_in[5];
  const float* kvnb = (const float*)d_in[6];
  const float* kvbw = (const float*)d_in[7];
  const float* ow   = (const float*)d_in[8];
  const float* fnw  = (const float*)d_in[9];
  const float* fnb  = (const float*)d_in[10];
  const float* c1w  = (const float*)d_in[11];
  const float* c1b  = (const float*)d_in[12];
  const float* c2w  = (const float*)d_in[13];
  const float* c2b  = (const float*)d_in[14];

  size_t off = 0;
  auto alloc = [&](size_t bytes) -> void* {
    off = (off + 255) & ~(size_t)255;
    void* p = (char*)d_ws + off;
    off += bytes;
    return p;
  };
  const size_t NTOK = (size_t)BB * TT;  // 4096 rows

  float*          x      = (float*)alloc(NTOK * HIDN * 4);
  unsigned short* hbf    = (unsigned short*)alloc(NTOK * HIDN * 2);
  float*          qf     = (float*)alloc(NTOK * HIDN * 4);
  unsigned short* qbf    = (unsigned short*)alloc(NTOK * HIDN * 2);
  float*          kvc    = (float*)alloc(NTOK * 320 * 4);
  unsigned short* latbf  = (unsigned short*)alloc(NTOK * RANKN * 2);
  float*          kvf    = (float*)alloc(NTOK * HIDN * 4);
  unsigned short* kbf    = (unsigned short*)alloc((size_t)BB * NKVN * TT * HDN * 2);
  unsigned short* vbfT   = (unsigned short*)alloc((size_t)BB * NKVN * HDN * TT * 2);
  float*          S      = (float*)alloc((size_t)TT * TT * 4);
  unsigned short* P      = (unsigned short*)alloc((size_t)TT * TT * 2);
  unsigned short* attnbf = (unsigned short*)alloc(NTOK * HIDN * 2);
  float*          proj   = (float*)alloc(NTOK * HIDN * 4);
  unsigned short* padh   = (unsigned short*)alloc((size_t)BB * TP * HIDN * 2);
  float*          ff1    = (float*)alloc(NTOK * FFN * 4);
  unsigned short* ff1p   = (unsigned short*)alloc((size_t)BB * TP * FFN * 2);
  float*          ff2    = (float*)alloc(NTOK * HIDN * 4);
  unsigned short* wtq    = (unsigned short*)alloc((size_t)HIDN * HIDN * 2);
  unsigned short* wtkva  = (unsigned short*)alloc((size_t)320 * HIDN * 2);
  unsigned short* wtkvb  = (unsigned short*)alloc((size_t)HIDN * RANKN * 2);
  unsigned short* wto    = (unsigned short*)alloc((size_t)HIDN * HIDN * 2);
  unsigned short* wtc1   = (unsigned short*)alloc((size_t)KTAP * FFN * HIDN * 2);
  unsigned short* wtc2   = (unsigned short*)alloc((size_t)KTAP * HIDN * FFN * 2);

  (void)hipMemcpyAsync(x, in_x, NTOK * HIDN * 4, hipMemcpyDeviceToDevice, stream);

  const float scale = 0.125f;  // HD^-0.5

  for (int li = 0; li < 2; ++li) {
    // ---- per-layer weight prep: cast (projections) / per-tap repack (convs) ----
    {
      int n = HIDN * HIDN;
      k_cast_bf16<<<(n + 255) / 256, 256, 0, stream>>>(qw + (size_t)li * n, wtq, n);
      n = 320 * HIDN;
      k_cast_bf16<<<(n + 255) / 256, 256, 0, stream>>>(kvaw + (size_t)li * n, wtkva, n);
      n = HIDN * RANKN;
      k_cast_bf16<<<(n + 255) / 256, 256, 0, stream>>>(kvbw + (size_t)li * n, wtkvb, n);
      n = HIDN * HIDN;
      k_cast_bf16<<<(n + 255) / 256, 256, 0, stream>>>(ow + (size_t)li * n, wto, n);
      const long nc = (long)FFN * HIDN * KTAP;
      k_conv_repack<<<(unsigned)((nc + 255) / 256), 256, 0, stream>>>(
          c1w + (size_t)li * nc, wtc1, FFN, HIDN, KTAP);
      k_conv_repack<<<(unsigned)((nc + 255) / 256), 256, 0, stream>>>(
          c2w + (size_t)li * nc, wtc2, HIDN, FFN, KTAP);
    }

    // ---- attention half ----
    k_layernorm_cast<<<(unsigned)NTOK, 256, 0, stream>>>(
        x, HIDN, HIDN, anw + li * HIDN, anb + li * HIDN, hbf, HIDN);

    gemm(stream, hbf, HIDN, wtq, HIDN, qf, HIDN, (int)NTOK, HIDN, HIDN, 1.f, 0.f, 0);
    gemm(stream, hbf, HIDN, wtkva, HIDN, kvc, 320, (int)NTOK, 320, HIDN, 1.f, 0.f, 0);

    k_layernorm_cast<<<(unsigned)NTOK, 256, 0, stream>>>(
        kvc, 320, RANKN, kvnw + li * RANKN, kvnb + li * RANKN, latbf, RANKN);

    gemm(stream, latbf, RANKN, wtkvb, RANKN, kvf, HIDN, (int)NTOK, HIDN, RANKN, 1.f, 0.f, 0);

    { int n = BB * TT * NHN * (HDN / 2);
      k_rope_pack_q<<<(n + 255) / 256, 256, 0, stream>>>(qf, qbf); }
    { int n = BB * TT * NKVN * (HDN / 2);
      k_rope_pack_kv<<<(n + 255) / 256, 256, 0, stream>>>(kvf, kbf, vbfT); }

    for (int b = 0; b < BB; ++b)
      for (int hk = 0; hk < NKVN; ++hk)
        for (int g = 0; g < GN; ++g) {
          const int hf = hk * GN + g;
          const unsigned short* Q  = qbf  + (((size_t)(b * NHN  + hf)) * TT)  * HDN;
          const unsigned short* Kn = kbf  + (((size_t)(b * NKVN + hk)) * TT)  * HDN;
          const unsigned short* Vt = vbfT + (((size_t)(b * NKVN + hk)) * HDN) * TT;
          // S = Q (T x HD) @ K^T     : BT = key (N=t', K=d) natural layout
          gemm(stream, Q, HDN, Kn, HDN, S, TT, TT, TT, HDN, 1.f, 0.f, 0);
          k_softmax_bf16<<<TT, 256, 0, stream>>>(S, P, TT, scale);
          // O = P (T x T) @ V        : BT = V^T (N=d, K=t')
          unsigned short* Co = attnbf + (size_t)(b * TT) * HIDN + hf * HDN;
          gemm(stream, P, TT, Vt, TT, Co, HIDN, TT, HDN, TT, 1.f, 0.f, 1);
        }

    gemm(stream, attnbf, HIDN, wto, HIDN, proj, HIDN, (int)NTOK, HIDN, HIDN, 1.f, 0.f, 0);
    { int n = (int)(NTOK * HIDN);
      k_add_res<<<(n + 255) / 256, 256, 0, stream>>>(x, proj, n); }

    // ---- conv FFN half ----
    k_layernorm_cast<<<(unsigned)NTOK, 256, 0, stream>>>(
        x, HIDN, HIDN, fnw + li * HIDN, fnb + li * HIDN, hbf, HIDN);
    { long n = (long)BB * TP * HIDN;
      k_pad_bf16<<<(unsigned)((n + 255) / 256), 256, 0, stream>>>(hbf, padh, HIDN); }

    // conv1 as 9 shifted accumulating GEMMs per batch: (TxHID)@(HIDxFF)
    for (int b = 0; b < BB; ++b)
      for (int k = 0; k < KTAP; ++k) {
        const unsigned short* Ab = padh + ((size_t)(b * TP + k)) * HIDN;
        const unsigned short* Bk = wtc1 + (size_t)k * FFN * HIDN;
        float* Cb = ff1 + (size_t)(b * TT) * FFN;
        gemm(stream, Ab, HIDN, Bk, HIDN, Cb, FFN, TT, FFN, HIDN,
             1.f, (k == 0) ? 0.f : 1.f, 0);
      }
    { long n = (long)BB * TP * FFN;
      k_relu_bias_pad<<<(unsigned)((n + 255) / 256), 256, 0, stream>>>(
          ff1, c1b + (size_t)li * FFN, ff1p); }

    // conv2 as 9 shifted accumulating GEMMs per batch: (TxFF)@(FFxHID)
    for (int b = 0; b < BB; ++b)
      for (int k = 0; k < KTAP; ++k) {
        const unsigned short* Ab = ff1p + ((size_t)(b * TP + k)) * FFN;
        const unsigned short* Bk = wtc2 + (size_t)k * HIDN * FFN;
        float* Cb = ff2 + (size_t)(b * TT) * HIDN;
        gemm(stream, Ab, FFN, Bk, FFN, Cb, HIDN, TT, HIDN, FFN,
             1.f, (k == 0) ? 0.f : 1.f, 0);
      }
    { int n = (int)(NTOK * HIDN);
      k_bias_add_res<<<(n + 255) / 256, 256, 0, stream>>>(
          x, ff2, c2b + (size_t)li * HIDN, n); }
  }

  (void)hipMemcpyAsync(d_out, x, NTOK * HIDN * 4, hipMemcpyDeviceToDevice, stream);
}
